// PoseEncoder_71665824301149
// MI455X (gfx1250) — matrix operand order repaired and checked by reading, not compiled, not run on hardware
//
#include <hip/hip_runtime.h>
#include <hip/hip_bf16.h>
#include <math.h>

// ---------------------------------------------------------------------------
// PoseEncoder for MI455X (gfx1250, wave32, WMMA).
//
// Key algebraic simplification: sequence length after peak-picking is 1, so
// softmax over a single key is identically 1 and attention output == V.
// Wq*/Wk* are never touched. The network reduces to a GEMM chain on an
// (8192 x feat) matrix, executed with v_wmma_f32_16x16x32_f16.
// ---------------------------------------------------------------------------

typedef __attribute__((ext_vector_type(16))) _Float16 v16h;
typedef __attribute__((ext_vector_type(8)))  _Float16 v8h;
typedef __attribute__((ext_vector_type(8)))  float    v8f;

#define BATCH 8192
#define TKP   305      // keypoint frames
#define TSEQ  303      // feature length after two diffs
#define K0    320      // TSEQ padded to multiple of 32
#define HID   768
#define NBINS 10

// ---------------------------------------------------------------------------
// Block-wide reduction helper (320 threads = 10 wave32s).
// ---------------------------------------------------------------------------
__device__ __forceinline__ float breduce(float v, bool domax, float* scr) {
#pragma unroll
  for (int o = 16; o > 0; o >>= 1) {
    float o2 = __shfl_down(v, o, 32);
    v = domax ? fmaxf(v, o2) : (v + o2);
  }
  const int w = threadIdx.x >> 5, l = threadIdx.x & 31;
  if (l == 0) scr[w] = v;
  __syncthreads();
  if (w == 0) {
    float r = (l < 10) ? scr[l] : (domax ? -3.0e38f : 0.0f);
#pragma unroll
    for (int o = 8; o > 0; o >>= 1) {
      float o2 = __shfl_down(r, o, 32);
      r = domax ? fmaxf(r, o2) : (r + o2);
    }
    if (l == 0) scr[0] = r;
  }
  __syncthreads();
  float res = scr[0];
  __syncthreads();
  return res;
}

// ---------------------------------------------------------------------------
// Fused preprocessing: directogram -> onset envelope -> peak pick ->
// (+pos_table) -> layernorm -> f16 activation row (K padded to 320).
// One block (320 threads) per batch element.
// ---------------------------------------------------------------------------
__global__ __launch_bounds__(320) void pre_kernel(
    const float* __restrict__ kp, const float* __restrict__ pos,
    const float* __restrict__ ln_g, const float* __restrict__ ln_b,
    _Float16* __restrict__ A0) {
  __shared__ float d[304 * NBINS];
  __shared__ float env[304];
  __shared__ float pk[304];
  __shared__ float scr[16];

  const int b = blockIdx.x;
  const int t = threadIdx.x;
  const float* kpb = kp + (size_t)b * TKP * 17 * 3;

  // 1) directogram row t (motion magnitude binned by phase)
  if (t < 304) {
    float bins[NBINS];
#pragma unroll
    for (int i = 0; i < NBINS; ++i) bins[i] = 0.0f;
    for (int j = 0; j < 17; ++j) {
      const float* p0 = kpb + ((size_t)t * 17 + j) * 3;
      const float* p1 = kpb + ((size_t)(t + 1) * 17 + j) * 3;
      float gx = p1[0] - p0[0];
      float gy = p1[1] - p0[1];
      float mag = sqrtf(gx * gx + gy * gy);
      float deg = atan2f(gy, gx) * 57.29577951308232f;   // rad -> deg
      deg = fmodf(deg, 180.0f);
      if (deg < 0.0f) deg += 180.0f;                     // python-style mod
      int pb = ((int)floorf(deg)) % NBINS;               // non-negative
      bins[pb] += mag;
    }
#pragma unroll
    for (int i = 0; i < NBINS; ++i) d[t * NBINS + i] = bins[i];
  }
  __syncthreads();

  // 2) spectral flux -> onset envelope
  if (t < TSEQ) {
    float e = 0.0f;
#pragma unroll
    for (int i = 0; i < NBINS; ++i) {
      float df = d[(t + 1) * NBINS + i] - d[t * NBINS + i];
      e += fmaxf(df, 0.0f);
    }
    env[t] = e;
  }
  if (t == TSEQ) env[TSEQ] = 0.0f;
  __syncthreads();

  // 3) normalize by global max
  float mx = breduce((t < TSEQ) ? env[t] : -3.0e38f, true, scr);
  if (t < TSEQ) env[t] = env[t] / mx;
  __syncthreads();

  // 4) peak picking
  float gsum = breduce((t < TSEQ) ? env[t] : 0.0f, false, scr);
  float gmean = gsum * (1.0f / TSEQ);
  float lm = 0.0f, lx = 0.0f;
  if (t < TSEQ - 3) {  // mean window 4, valid length 300, zero-padded
    lm = (env[t] + env[t + 1] + env[t + 2] + env[t + 3]) * 0.25f;
  }
  if (t < TSEQ - 5) {  // max window 6, valid length 298, zero-padded
    lx = env[t];
#pragma unroll
    for (int k = 1; k < 6; ++k) lx = fmaxf(lx, env[t + k]);
  }
  float p = 0.0f;
  if (t < TSEQ) p = ((lx - lm > 0.1f * gmean) && (lx == env[t])) ? 1.0f : 0.0f;
  if (t < TSEQ) pk[t] = p;
  if (t == TSEQ) pk[TSEQ] = 0.0f;
  __syncthreads();
  float fin = 0.0f;
  if (t < TSEQ - 1) fin = (pk[t + 1] != pk[t]) ? pk[t] : 0.0f;  // edge mask

  // 5) + positional table, layernorm over 303 features
  float xv = (t < TSEQ) ? (fin + pos[t]) : 0.0f;
  float s1 = breduce((t < TSEQ) ? xv : 0.0f, false, scr);
  float mean = s1 * (1.0f / TSEQ);
  float dv = (t < TSEQ) ? (xv - mean) : 0.0f;
  float s2 = breduce(dv * dv, false, scr);
  float rstd = rsqrtf(s2 * (1.0f / TSEQ) + 1e-5f);

  if (t < TSEQ) {
    A0[(size_t)b * K0 + t] = (_Float16)(dv * rstd * ln_g[t] + ln_b[t]);
  } else if (t < K0) {
    A0[(size_t)b * K0 + t] = (_Float16)0.0f;  // zero pad K to 320
  }
}

// ---------------------------------------------------------------------------
// f32 -> f16 weight conversion with zero K-padding.
// ---------------------------------------------------------------------------
__global__ void wconv(const float* __restrict__ src, _Float16* __restrict__ dst,
                      int N, int Ks, int Kd) {
  int i = blockIdx.x * blockDim.x + threadIdx.x;
  if (i >= N * Kd) return;
  int n = i / Kd, k = i - n * Kd;
  dst[i] = (_Float16)((k < Ks) ? src[(size_t)n * Ks + k] : 0.0f);
}

// ---------------------------------------------------------------------------
// WMMA GEMM: Out = epi(A(MxK,f16) @ W(NxK,f16)^T + bias), row-major, K|32.
// Block = 256 threads = 8 waves; block tile 128x128; wave tile 32x64
// (2x4 tiles of v_wmma_f32_16x16x32_f16). Fragments load directly from
// global: W rows are K-contiguous == this code's B k order (k = 16h + s, same as A); weights
// are fully L2-resident (192 MB L2).
// MODE: 0 = plain->f16, 1 = silu->f16, 2 = +resid->f16, 3 = plain->f32.
// ---------------------------------------------------------------------------
template <int MODE>
__global__ __launch_bounds__(256) void gemm_wmma(
    const _Float16* __restrict__ A, const _Float16* __restrict__ W,
    const float* __restrict__ bias, const _Float16* __restrict__ resid,
    void* __restrict__ Out, int M, int N, int K) {
  const int tid  = threadIdx.x;
  const int lane = tid & 31;
  const int wave = tid >> 5;
  const int wm   = wave & 3;                 // 4 waves along M
  const int wn   = wave >> 2;                // 2 waves along N
  const int mb   = blockIdx.x * 128 + wm * 32;
  const int nb   = blockIdx.y * 128 + wn * 64;
  const int row  = lane & 15;
  const int hi   = lane >> 4;

  v8f acc[2][4];
#pragma unroll
  for (int mt = 0; mt < 2; ++mt)
#pragma unroll
    for (int nt = 0; nt < 4; ++nt)
#pragma unroll
      for (int r = 0; r < 8; ++r) acc[mt][nt][r] = 0.0f;

  // A-fragment 16x32 f16 layout: lanes 0-15 rows M=0..15 hold K 0..15,
  // lanes 16-31 hold K 16..31  -> two v8h loads per tile per lane.
  const _Float16* Ar[2];
  Ar[0] = A + (size_t)(mb + row) * K + 16 * hi;
  Ar[1] = A + (size_t)(mb + 16 + row) * K + 16 * hi;
  // B-fragment 32x16 f16: lane = column n, lanes 0-15 hold K 0..15,
  // lanes 16-31 hold K 16..31 -> one v16h row slice of W per tile per lane.
  const _Float16* Br[4];
#pragma unroll
  for (int nt = 0; nt < 4; ++nt)
    Br[nt] = W + (size_t)(nb + nt * 16 + row) * K + 16 * hi;

  for (int k = 0; k < K; k += 32) {
    if (k + 32 < K) {  // prefetch next A k-tile (global_prefetch_b8)
      __builtin_prefetch(Ar[0] + k + 32, 0, 1);
      __builtin_prefetch(Ar[1] + k + 32, 0, 1);
    }
    v16h af[2];
#pragma unroll
    for (int mt = 0; mt < 2; ++mt) {
      v8h lo = *(const v8h*)(Ar[mt] + k);
      v8h hh = *(const v8h*)(Ar[mt] + k + 8);
      af[mt] = __builtin_shufflevector(lo, hh, 0, 1, 2, 3, 4, 5, 6, 7, 8, 9,
                                       10, 11, 12, 13, 14, 15);
    }
    v16h bf[4];
#pragma unroll
    for (int nt = 0; nt < 4; ++nt) bf[nt] = *(const v16h*)(Br[nt] + k);
#pragma unroll
    for (int mt = 0; mt < 2; ++mt)
#pragma unroll
      for (int nt = 0; nt < 4; ++nt)
        acc[mt][nt] = __builtin_amdgcn_wmma_f32_16x16x32_f16(
            false, af[mt], false, bf[nt], (short)0, acc[mt][nt], false, false);
  }

  // Epilogue. C/D layout: lane L -> column n = L%16; VGPR r -> row r + 8*(L/16).
#pragma unroll
  for (int nt = 0; nt < 4; ++nt) {
    const int n = nb + nt * 16 + row;
    const float bs = bias ? bias[n] : 0.0f;
#pragma unroll
    for (int mt = 0; mt < 2; ++mt) {
#pragma unroll
      for (int r = 0; r < 8; ++r) {
        const int m = mb + mt * 16 + r + 8 * hi;
        float val = acc[mt][nt][r] + bs;
        if (MODE == 1) val = val / (1.0f + expf(-val));          // SiLU
        if (MODE == 2) val += (float)resid[(size_t)m * N + n];   // residual
        if (MODE == 3)
          ((float*)Out)[(size_t)m * N + n] = val;
        else
          ((_Float16*)Out)[(size_t)m * N + n] = (_Float16)val;
      }
    }
  }
}

// ---------------------------------------------------------------------------
// Host-side orchestration.
// ---------------------------------------------------------------------------
extern "C" void kernel_launch(void* const* d_in, const int* in_sizes, int n_in,
                              void* d_out, int out_size, void* d_ws,
                              size_t ws_size, hipStream_t stream) {
  (void)in_sizes; (void)n_in; (void)out_size; (void)ws_size;
  const float* kp   = (const float*)d_in[0];
  const float* pos  = (const float*)d_in[1];
  const float* ln_g = (const float*)d_in[2];
  const float* ln_b = (const float*)d_in[3];
  const float* W1   = (const float*)d_in[4];
  const float* b1   = (const float*)d_in[5];
  const float* W2   = (const float*)d_in[6];
  const float* b2   = (const float*)d_in[7];
  // d_in[8], d_in[9]  = Wq1, Wk1 : dead (softmax over 1 key == 1)
  const float* Wv1  = (const float*)d_in[10];
  const float* Wo1  = (const float*)d_in[11];
  const float* bo1  = (const float*)d_in[12];
  // d_in[13], d_in[14] = Wq2, Wk2 : dead
  const float* Wv2  = (const float*)d_in[15];
  const float* Wo2  = (const float*)d_in[16];
  const float* bo2  = (const float*)d_in[17];
  const float* Wn   = (const float*)d_in[18];
  const float* bn   = (const float*)d_in[19];

  // Workspace layout (bytes, all 32B-aligned).
  char* ws = (char*)d_ws;
  const size_t SZ_A0  = (size_t)BATCH * K0 * 2;        // 5.24 MB
  const size_t SZ_W1  = (size_t)HID * K0 * 2;          // 0.49 MB
  const size_t SZ_W   = (size_t)HID * HID * 2;         // 1.18 MB each
  const size_t SZ_H   = (size_t)BATCH * HID * 2;       // 12.58 MB each
  size_t off = 0;
  _Float16* A0   = (_Float16*)(ws + off); off += SZ_A0;
  _Float16* W1h  = (_Float16*)(ws + off); off += SZ_W1;
  _Float16* W2h  = (_Float16*)(ws + off); off += SZ_W;
  _Float16* Wv1h = (_Float16*)(ws + off); off += SZ_W;
  _Float16* Wo1h = (_Float16*)(ws + off); off += SZ_W;
  _Float16* Wv2h = (_Float16*)(ws + off); off += SZ_W;
  _Float16* Wo2h = (_Float16*)(ws + off); off += SZ_W;
  _Float16* Wnh  = (_Float16*)(ws + off); off += SZ_W;
  _Float16* BufA = (_Float16*)(ws + off); off += SZ_H;  // H1 / V1 / V2
  _Float16* X    = (_Float16*)(ws + off); off += SZ_H;  // x  (then reused as x2)
  _Float16* X1   = (_Float16*)(ws + off); off += SZ_H;  // x1
  // total ~50.6 MB

  // 1) fused preprocessing -> A0 (8192 x 320 f16)
  pre_kernel<<<BATCH, 320, 0, stream>>>(kp, pos, ln_g, ln_b, A0);

  // 2) weight conversion (L2-resident, trivially cheap)
  wconv<<<(HID * K0 + 255) / 256, 256, 0, stream>>>(W1, W1h, HID, TSEQ, K0);
  const int g768 = (HID * HID + 255) / 256;
  wconv<<<g768, 256, 0, stream>>>(W2,  W2h,  HID, HID, HID);
  wconv<<<g768, 256, 0, stream>>>(Wv1, Wv1h, HID, HID, HID);
  wconv<<<g768, 256, 0, stream>>>(Wo1, Wo1h, HID, HID, HID);
  wconv<<<g768, 256, 0, stream>>>(Wv2, Wv2h, HID, HID, HID);
  wconv<<<g768, 256, 0, stream>>>(Wo2, Wo2h, HID, HID, HID);
  wconv<<<g768, 256, 0, stream>>>(Wn,  Wnh,  HID, HID, HID);

  // 3) GEMM chain (grid: M/128 x N/128 = 64 x 6)
  dim3 grid(BATCH / 128, HID / 128), blk(256);
  // H1 = silu(A0 @ W1^T + b1)
  gemm_wmma<1><<<grid, blk, 0, stream>>>(A0, W1h, b1, nullptr, BufA,
                                         BATCH, HID, K0);
  // X = H1 @ W2^T + b2
  gemm_wmma<0><<<grid, blk, 0, stream>>>(BufA, W2h, b2, nullptr, X,
                                         BATCH, HID, HID);
  // V1 = X @ Wv1^T           (attention 1 collapses to V)
  gemm_wmma<0><<<grid, blk, 0, stream>>>(X, Wv1h, nullptr, nullptr, BufA,
                                         BATCH, HID, HID);
  // X1 = V1 @ Wo1^T + bo1 + X
  gemm_wmma<2><<<grid, blk, 0, stream>>>(BufA, Wo1h, bo1, X, X1,
                                         BATCH, HID, HID);
  // V2 = X @ Wv2^T           (attention 2: ctx is still X)
  gemm_wmma<0><<<grid, blk, 0, stream>>>(X, Wv2h, nullptr, nullptr, BufA,
                                         BATCH, HID, HID);
  // X2 = V2 @ Wo2^T + bo2 + X1   (overwrites X, which is no longer needed)
  gemm_wmma<2><<<grid, blk, 0, stream>>>(BufA, Wo2h, bo2, X1, X,
                                         BATCH, HID, HID);
  // out = X2 @ Wn^T + bn  (f32)
  gemm_wmma<3><<<grid, blk, 0, stream>>>(X, Wnh, bn, nullptr, d_out,
                                         BATCH, HID, HID);
}
